// Pddm_61057255080329
// MI455X (gfx1250) — matrix operand order repaired and checked by reading, not compiled, not run on hardware
//
#include <hip/hip_runtime.h>
#include <math.h>

#define NPT 512
#define DD  128

typedef __attribute__((ext_vector_type(16))) __bf16 v16bf;
typedef __attribute__((ext_vector_type(8)))  float  v8f;

union FragBF { v16bf v; __bf16 e[16]; };

// ---------------------------------------------------------------- kernel 0
__global__ void cvt_weights(const float* __restrict__ wu, const float* __restrict__ wv,
                            const float* __restrict__ wc,
                            __bf16* __restrict__ wubf, __bf16* __restrict__ wvbf,
                            __bf16* __restrict__ wcbf) {
    int idx = blockIdx.x * blockDim.x + threadIdx.x;   // 32768 threads
    if (idx < DD * DD) {
        wubf[idx] = (__bf16)wu[idx];
        wvbf[idx] = (__bf16)wv[idx];
    }
    if (idx < DD * 2 * DD) wcbf[idx] = (__bf16)wc[idx];
}

// ---------------------------------------------------------------- kernel 1
// First-layer GEMM: relu(A(16x128) * W^T + b). Stores UNNORMALIZED bf16
// activations to LDS; returns per-row sum-of-squares (rows r+8g) in registers.
__device__ __forceinline__ void gemm_layer(const FragBF (&A)[4],
                                           const __bf16* W, const float* bias,
                                           __bf16* dst, float (&rs)[8],
                                           int g, int nl) {
#pragma unroll
    for (int r = 0; r < 8; ++r) rs[r] = 0.f;
    for (int t = 0; t < 8; ++t) {                 // 8 N-tiles of 16
        float bn = bias[t * 16 + nl];
        v8f c;
#pragma unroll
        for (int r = 0; r < 8; ++r) c[r] = bn;
#pragma unroll
        for (int kk = 0; kk < 4; ++kk) {          // K = 128 in 4 steps of 32
            FragBF b;                             // B[k][n] = W[n][k], from LDS
            const __bf16* wp = W + (t * 16 + nl) * DD + kk * 32 + 16 * g;
#pragma unroll
            for (int h = 0; h < 16; ++h) b.e[h] = wp[h];
            c = __builtin_amdgcn_wmma_f32_16x16x32_bf16(false, A[kk].v, false, b.v,
                                                        (short)0, c, false, false);
        }
#pragma unroll
        for (int r = 0; r < 8; ++r) {
            float x = fmaxf(c[r], 0.f);
            rs[r] += x * x;
            dst[(r + 8 * g) * DD + t * 16 + nl] = (__bf16)x;
        }
    }
#pragma unroll
    for (int r = 0; r < 8; ++r) {
#pragma unroll
        for (int m = 1; m < 16; m <<= 1) rs[r] += __shfl_xor(rs[r], m, 32);
    }
}

__launch_bounds__(128, 1)
__global__ void pair_mlp(const float* __restrict__ F,
                         const float* __restrict__ bu, const float* __restrict__ bv,
                         const float* __restrict__ bc,
                         const float* __restrict__ wsw, const float* __restrict__ wsb,
                         const __bf16* __restrict__ wAll,   // Wu|Wv|Wc bf16, 128 KB
                         float* __restrict__ sOut) {
    // dynamic LDS: Wu(32K) | Wv(32K) | Wc(64K) | uAct(16K) | vAct(16K) = 160 KB
    extern __shared__ char smem[];
    __bf16* sWu  = (__bf16*)smem;
    __bf16* sWv  = sWu + DD * DD;
    __bf16* sWc  = sWv + DD * DD;
    __bf16* sAct = sWc + DD * 2 * DD;

    const int wave = threadIdx.x >> 5;
    const int lane = threadIdx.x & 31;
    const int g  = lane >> 4;       // half-wave group
    const int nl = lane & 15;       // A-row m / B-col n within tile

    // ---- stage all weights global->LDS with CDNA5 async copies (in flight
    // while we build the A-fragments from features below).
    {
        unsigned ldsW = (unsigned)(uintptr_t)(void*)sWu;  // low 32b = LDS offset
        for (int it = 0; it < 64; ++it) {                 // 128 thr * 16 B * 64
            unsigned off = (unsigned)(threadIdx.x * 16 + it * 2048);
            unsigned la  = ldsW + off;
            asm volatile("global_load_async_to_lds_b128 %0, %1, %2"
                         :: "v"(la), "v"(off), "s"((const void*)wAll) : "memory");
        }
    }

    __bf16* uL = sAct + wave * (16 * DD);
    __bf16* vL = sAct + 4 * (16 * DD) + wave * (16 * DD);

    const int a  = blockIdx.y;
    const int b0 = blockIdx.x * 64 + wave * 16;
    const float* fa = F + a * DD;
    const float* fb = F + (b0 + nl) * DD;

    // Build bf16 A-fragments for u = |fa-fb|, v = 0.5*(fa+fb).
    // 16-bit A 16x32 layout (contiguous, matches B): K = 16*grp + h.
    FragBF Au[4], Av[4];
#pragma unroll
    for (int kk = 0; kk < 4; ++kk) {
#pragma unroll
        for (int h = 0; h < 16; ++h) {
            int k = kk * 32 + 16 * g + h;
            float xa = fa[k], xb = fb[k];
            Au[kk].e[h] = (__bf16)fabsf(xa - xb);
            Av[kk].e[h] = (__bf16)(0.5f * (xa + xb));
        }
    }

    asm volatile("s_wait_asynccnt 0x0" ::: "memory");
    __syncthreads();   // weights visible to all waves

    float rsU[8], rsV[8];
    gemm_layer(Au, sWu, bu, uL, rsU, g, nl);
    gemm_layer(Av, sWv, bv, vL, rsV, g, nl);

    // l2norm scales are per-row => fold AFTER the concat WMMAs:
    // h = relu(su[m]*(u_relu . WcU) + sv[m]*(v_relu . WcV) + bc)
    float suR[8], svR[8], dacc[8];
#pragma unroll
    for (int r = 0; r < 8; ++r) {
        suR[r] = 1.f / fmaxf(sqrtf(rsU[r]), 1e-12f);
        svR[r] = 1.f / fmaxf(sqrtf(rsV[r]), 1e-12f);
        dacc[r] = 0.f;
    }

    for (int t = 0; t < 8; ++t) {
        v8f cU, cV;
#pragma unroll
        for (int r = 0; r < 8; ++r) { cU[r] = 0.f; cV[r] = 0.f; }
#pragma unroll
        for (int kk = 0; kk < 4; ++kk) {          // u half: Wc cols 0..127
            FragBF afr, bfr;
#pragma unroll
            for (int h = 0; h < 16; ++h) {
                int k = kk * 32 + 16 * g + h;
                afr.e[h] = uL[nl * DD + k];       // raw bf16 copy, no cvt
            }
            const __bf16* wp = sWc + (t * 16 + nl) * (2 * DD) + kk * 32 + 16 * g;
#pragma unroll
            for (int h = 0; h < 16; ++h) bfr.e[h] = wp[h];
            cU = __builtin_amdgcn_wmma_f32_16x16x32_bf16(false, afr.v, false, bfr.v,
                                                         (short)0, cU, false, false);
        }
#pragma unroll
        for (int kk = 0; kk < 4; ++kk) {          // v half: Wc cols 128..255
            FragBF afr, bfr;
#pragma unroll
            for (int h = 0; h < 16; ++h) {
                int k = kk * 32 + 16 * g + h;
                afr.e[h] = vL[nl * DD + k];
            }
            const __bf16* wp = sWc + (t * 16 + nl) * (2 * DD) + DD + kk * 32 + 16 * g;
#pragma unroll
            for (int h = 0; h < 16; ++h) bfr.e[h] = wp[h];
            cV = __builtin_amdgcn_wmma_f32_16x16x32_bf16(false, afr.v, false, bfr.v,
                                                         (short)0, cV, false, false);
        }
        float bn  = bc[t * 16 + nl];
        float wsn = wsw[t * 16 + nl];
#pragma unroll
        for (int r = 0; r < 8; ++r)
            dacc[r] += fmaxf(suR[r] * cU[r] + svR[r] * cV[r] + bn, 0.f) * wsn;
    }
#pragma unroll
    for (int r = 0; r < 8; ++r) {
#pragma unroll
        for (int m = 1; m < 16; m <<= 1) dacc[r] += __shfl_xor(dacc[r], m, 32);
    }
    if (nl == 0) {
        float bsv = wsb[0];
#pragma unroll
        for (int r = 0; r < 8; ++r)
            sOut[a * NPT + b0 + 8 * g + r] = dacc[r] + bsv;
    }
}

// ---------------------------------------------------------------- kernel 2
__launch_bounds__(1024, 1)
__global__ void finalize(const float* __restrict__ sBuf, const int* __restrict__ y,
                         const float* __restrict__ F, float* __restrict__ out) {
    __shared__ float rv[1024];
    __shared__ int   ri[1024];
    __shared__ int   s_ij, s_k, s_l;
    __shared__ float s_smin, s_smax;
    const int tid = threadIdx.x;
    const float PINF = __builtin_inff();

    // argmin over positive off-diagonal pairs (first-index tie-break)
    float best = PINF; int bi = 0x7FFFFFFF;
    for (int idx = tid; idx < NPT * NPT; idx += 1024) {
        int r = idx >> 9, c = idx & (NPT - 1);
        if (r != c && y[r] == y[c]) {
            float v = sBuf[idx];
            if (v < best || (v == best && idx < bi)) { best = v; bi = idx; }
        }
    }
    rv[tid] = best; ri[tid] = bi;
    __syncthreads();
    for (int off = 512; off > 0; off >>= 1) {
        if (tid < off) {
            float v = rv[tid + off]; int ii = ri[tid + off];
            if (v < rv[tid] || (v == rv[tid] && ii < ri[tid])) { rv[tid] = v; ri[tid] = ii; }
        }
        __syncthreads();
    }
    if (tid == 0) s_ij = ri[0];
    __syncthreads();
    const int i = s_ij >> 9, j = s_ij & (NPT - 1);

    float sneg_i = -PINF, sneg_j = -PINF, rmin = PINF, rmax = -PINF;
    if (tid < NPT) {
        float si = sBuf[i * NPT + tid];
        float sj = sBuf[j * NPT + tid];
        sneg_i = (y[i] == y[tid]) ? 0.f : si;
        sneg_j = (y[j] == y[tid]) ? 0.f : sj;
        rmin = fminf(sneg_i, sneg_j);
        rmax = fmaxf(sneg_i, sneg_j);
    }
    // argmax k over sneg_i
    rv[tid] = sneg_i; ri[tid] = (tid < NPT) ? tid : 0x7FFFFFFF;
    __syncthreads();
    for (int off = 512; off > 0; off >>= 1) {
        if (tid < off) {
            float v = rv[tid + off]; int ii = ri[tid + off];
            if (v > rv[tid] || (v == rv[tid] && ii < ri[tid])) { rv[tid] = v; ri[tid] = ii; }
        }
        __syncthreads();
    }
    if (tid == 0) s_k = ri[0];
    __syncthreads();
    // argmax l over sneg_j
    rv[tid] = sneg_j; ri[tid] = (tid < NPT) ? tid : 0x7FFFFFFF;
    __syncthreads();
    for (int off = 512; off > 0; off >>= 1) {
        if (tid < off) {
            float v = rv[tid + off]; int ii = ri[tid + off];
            if (v > rv[tid] || (v == rv[tid] && ii < ri[tid])) { rv[tid] = v; ri[tid] = ii; }
        }
        __syncthreads();
    }
    if (tid == 0) s_l = ri[0];
    __syncthreads();
    // smin
    rv[tid] = rmin; __syncthreads();
    for (int off = 512; off > 0; off >>= 1) {
        if (tid < off) rv[tid] = fminf(rv[tid], rv[tid + off]);
        __syncthreads();
    }
    if (tid == 0) s_smin = rv[0];
    __syncthreads();
    // smax
    rv[tid] = rmax; __syncthreads();
    for (int off = 512; off > 0; off >>= 1) {
        if (tid < off) rv[tid] = fmaxf(rv[tid], rv[tid + off]);
        __syncthreads();
    }
    if (tid == 0) s_smax = rv[0];
    __syncthreads();

    if (tid == 0) {
        const int k = s_k, l = s_l;
        const float smin = s_smin, smax = s_smax;
        const float inv = 1.f / (smax - smin);
        float sij = sBuf[i * NPT + j];
        float sik = sBuf[i * NPT + k];
        float sjl = sBuf[j * NPT + l];
        float snij = (sij - smin) * inv;
        float snik = (sik - smin) * inv;
        float snjl = (sjl - smin) * inv;
        float Em = fmaxf(0.5f + snik - snij, 0.f) + fmaxf(0.5f + snjl - snij, 0.f);
        float dij = 0.f, dik = 0.f, djl = 0.f;
        for (int q = 0; q < DD; ++q) {
            float fi = F[i * DD + q], fj = F[j * DD + q];
            float fk = F[k * DD + q], fl = F[l * DD + q];
            float t1 = fi - fj; dij += t1 * t1;
            float t2 = fi - fk; dik += t2 * t2;
            float t3 = fj - fl; djl += t3 * t3;
        }
        float Ee = fmaxf(1.0f + dij - dik, 0.f) + fmaxf(1.0f + dij - djl, 0.f);
        out[0] = Em + 0.5f * Ee;
    }
}

// ---------------------------------------------------------------- launch
extern "C" void kernel_launch(void* const* d_in, const int* in_sizes, int n_in,
                              void* d_out, int out_size, void* d_ws, size_t ws_size,
                              hipStream_t stream) {
    const float* F    = (const float*)d_in[0];
    const int*   y    = (const int*)  d_in[1];
    const float* wu_w = (const float*)d_in[2];
    const float* wu_b = (const float*)d_in[3];
    const float* wv_w = (const float*)d_in[4];
    const float* wv_b = (const float*)d_in[5];
    const float* wc_w = (const float*)d_in[6];
    const float* wc_b = (const float*)d_in[7];
    const float* ws_w = (const float*)d_in[8];
    const float* ws_b = (const float*)d_in[9];
    float* out = (float*)d_out;

    float* sBuf = (float*)d_ws;                          // 512*512 f32 = 1 MB
    __bf16* wubf = (__bf16*)(sBuf + NPT * NPT);          // contiguous 128 KB:
    __bf16* wvbf = wubf + DD * DD;                       //   Wu | Wv | Wc
    __bf16* wcbf = wvbf + DD * DD;

    const int smemBytes = (2 * DD * DD + DD * 2 * DD + 2 * 4 * 16 * DD) * 2; // 160 KB
    (void)hipFuncSetAttribute((const void*)pair_mlp,
                              hipFuncAttributeMaxDynamicSharedMemorySize, smemBytes);

    cvt_weights<<<dim3(128), dim3(256), 0, stream>>>(wu_w, wv_w, wc_w, wubf, wvbf, wcbf);
    pair_mlp<<<dim3(8, NPT), dim3(128), smemBytes, stream>>>(F, wu_b, wv_b, wc_b,
                                                             ws_w, ws_b, wubf, sBuf);
    finalize<<<dim3(1), dim3(1024), 0, stream>>>(sBuf, y, F, out);
}
